// CFConvS2V_87445534146867
// MI455X (gfx1250) — matrix operand order repaired and checked by reading, not compiled, not run on hardware
//
#include <hip/hip_runtime.h>
#include <hip/hip_bf16.h>
#include <stdint.h>

// MI455X / gfx1250: wave32, WMMA 16x16x32 bf16 -> f32 accumulate.
typedef __attribute__((ext_vector_type(16))) __bf16 v16bf;
typedef __attribute__((ext_vector_type(8)))  float  v8f;
typedef __attribute__((ext_vector_type(4)))  float  f4;   // true vector type (NT-load legal)

#define B_ 32
#define N_ 512
#define H_ 128

#define WPITCH 132   // kernel A LDS row pitch (ushorts), multiple of 4
#define APITCH 36    // kernel B A-tile row pitch (ushorts), multiple of 4

// Hardware f32 -> bf16 (RNE) conversion.
__device__ __forceinline__ unsigned short f2bf(float f) {
    union { __bf16 h; unsigned short s; } u;
    u.h = (__bf16)f;
    return u.s;
}

union BF16Frag { uint32_t u[8]; v16bf v; };
union Pack4 { unsigned short sh[4]; uint64_t q; };

// ---------------------------------------------------------------------------
// Kernel A: h = SiLU(LayerNorm(s @ W1^T + b1)), emitted TRANSPOSED as
// h_t[b][d][j] (bf16) into d_ws so kernel B's B-fragments are contiguous.
// One block = 32 rows (within a single batch since 32 | 512). 256 thr = 8 waves.
// ---------------------------------------------------------------------------
__global__ __launch_bounds__(256)
void lin1_ln_silu_kernel(const float* __restrict__ s,
                         const float* __restrict__ W1,
                         const float* __restrict__ b1,
                         unsigned short* __restrict__ h_t) {
    // W1 row-major [k][h] is exactly B^T[k_out][K] -> contiguous K-pairs.
    __shared__ __attribute__((aligned(16))) unsigned short Wt[128 * WPITCH];
    __shared__ __attribute__((aligned(16))) unsigned short Sa[32 * WPITCH];
    __shared__ __attribute__((aligned(16))) float Hf[32 * 128];
    __shared__ float redS[32 * 8];
    __shared__ float redQ[32 * 8];
    __shared__ float stats[32 * 2];

    const int t    = threadIdx.x;
    const int row0 = blockIdx.x * 32;

    // Vectorized bf16 staging: float4 loads + b64 LDS stores.
    for (int e = t * 4; e < 128 * 128; e += 1024) {
        int k = e >> 7, h = e & 127;
        f4 v = *(const f4*)(W1 + e);
        Pack4 pk;
        pk.sh[0] = f2bf(v.x); pk.sh[1] = f2bf(v.y);
        pk.sh[2] = f2bf(v.z); pk.sh[3] = f2bf(v.w);
        *(uint64_t*)&Wt[k * WPITCH + h] = pk.q;
    }
    for (int e = t * 4; e < 32 * 128; e += 1024) {
        int r = e >> 7, h = e & 127;
        f4 v = *(const f4*)(s + (size_t)(row0 + r) * 128 + h);
        Pack4 pk;
        pk.sh[0] = f2bf(v.x); pk.sh[1] = f2bf(v.y);
        pk.sh[2] = f2bf(v.z); pk.sh[3] = f2bf(v.w);
        *(uint64_t*)&Sa[r * WPITCH + h] = pk.q;
    }
    __syncthreads();

    const int w     = t >> 5;
    const int l     = t & 31;
    const int half  = l >> 4;
    const int ln    = l & 15;
    const int m_sub = w & 1;
    const int ngrp  = w >> 1;

    v8f acc[2] = {};
#pragma unroll
    for (int kk = 0; kk < 4; ++kk) {
        // A fragment (16x32 bf16): lane row = m_sub*16 + ln; K split per half.
        BF16Frag a;
        const int r = m_sub * 16 + ln;
#pragma unroll
        for (int p = 0; p < 8; ++p) {
            int klocal = kk * 32 + half * 16 + 2 * p;
            a.u[p] = *(const uint32_t*)&Sa[r * WPITCH + klocal];
        }
#pragma unroll
        for (int dn = 0; dn < 2; ++dn) {
            // B fragment (32x16 bf16): lane col = ngrp*32+dn*16+ln; K half per lane-half.
            BF16Frag bfr;
            const int col = ngrp * 32 + dn * 16 + ln;
#pragma unroll
            for (int p = 0; p < 8; ++p) {
                int k = kk * 32 + half * 16 + 2 * p;
                bfr.u[p] = *(const uint32_t*)&Wt[col * WPITCH + k];
            }
            acc[dn] = __builtin_amdgcn_wmma_f32_16x16x32_bf16(
                false, a.v, false, bfr.v, (short)0, acc[dn], false, false);
        }
    }

    // C/D layout: lane l, VGPR p -> M = p + 8*(l>>4), N = l&15.
#pragma unroll
    for (int dn = 0; dn < 2; ++dn) {
        const int col  = ngrp * 32 + dn * 16 + ln;
        const float bb = b1[col];
#pragma unroll
        for (int p = 0; p < 8; ++p) {
            int m = m_sub * 16 + p + half * 8;
            Hf[m * 128 + col] = acc[dn][p] + bb;
        }
    }
    __syncthreads();

    // LayerNorm stats (no affine): 8 threads per row.
    {
        const int r = t >> 3, seg = t & 7;
        float sS = 0.f, sQ = 0.f;
#pragma unroll
        for (int i = 0; i < 16; ++i) {
            float x = Hf[r * 128 + seg * 16 + i];
            sS += x; sQ += x * x;
        }
        redS[r * 8 + seg] = sS;
        redQ[r * 8 + seg] = sQ;
    }
    __syncthreads();
    if (t < 32) {
        float sS = 0.f, sQ = 0.f;
#pragma unroll
        for (int i = 0; i < 8; ++i) { sS += redS[t * 8 + i]; sQ += redQ[t * 8 + i]; }
        float mu  = sS * (1.0f / 128.0f);
        float var = sQ * (1.0f / 128.0f) - mu * mu;
        stats[t * 2]     = mu;
        stats[t * 2 + 1] = rsqrtf(var + 1e-5f);
    }
    __syncthreads();

    // Normalize + SiLU + coalesced TRANSPOSED store: h_t[b][d][j0+rb .. +16]
    {
        const int b  = row0 >> 9;        // batch (32 | 512 => block within one batch)
        const int j0 = row0 & (N_ - 1);  // j offset inside batch
        const int d  = t >> 1;
        const int rb = (t & 1) * 16;
        union { unsigned short sh[16]; uint4 q[2]; } pk;
#pragma unroll
        for (int i = 0; i < 16; ++i) {
            int r = rb + i;
            float x = (Hf[r * 128 + d] - stats[r * 2]) * stats[r * 2 + 1];
            pk.sh[i] = f2bf(x / (1.0f + __expf(-x)));   // SiLU
        }
        uint4* dst = (uint4*)(h_t + (((size_t)b * H_ + d) * N_ + j0 + rb));
        dst[0] = pk.q[0];
        dst[1] = pk.q[1];
    }
}

// ---------------------------------------------------------------------------
// Kernel B: v[b,i,c,d] = sum_j (ev*mask)[b,i,j,c] * h[b,j,d]
// Block: (b, 32-row i tile). 8 waves; wave w -> m_sub=w&1, d-group=(w>>1)*32.
// K loop over j in steps of 32, DOUBLE-BUFFERED A tiles. Staging: each thread
// owns 4 consecutive j of one i-row -> 4x16B NT vector loads + 3x ds_store_b64.
// B fragments loaded straight from global (h_t is [b][d][j], so the 16
// K-values per lane are 32 contiguous bytes).
// ---------------------------------------------------------------------------
__global__ __launch_bounds__(256)
void cfconv_wmma_kernel(const float* __restrict__ ev,
                        const float* __restrict__ mask,
                        const unsigned short* __restrict__ h_t,
                        float* __restrict__ out) {
    __shared__ __attribute__((aligned(16))) unsigned short At[2 * 3 * 32 * APITCH];

    const int t  = threadIdx.x;
    const int b  = blockIdx.y;
    const int i0 = blockIdx.x * 32;

    const int w     = t >> 5;
    const int l     = t & 31;
    const int half  = l >> 4;
    const int ln    = l & 15;
    const int m_sub = w & 1;
    const int dgrp  = w >> 1;

    const size_t evRowBase = (size_t)(b * N_ + i0) * N_;

    auto stage = [&](int buf, int jj) {
        unsigned short* A = At + buf * (3 * 32 * APITCH);
        const int i  = t >> 3;            // one of 32 rows
        const int jb = (t & 7) * 4;       // 4 consecutive j
        size_t base = evRowBase + (size_t)i * N_ + jj + jb;
        // mask: 4 floats; ev: 12 interleaved floats [j][c] -> 3x float4
        f4 m4 = __builtin_nontemporal_load((const f4*)(mask + base));
        const f4* ep = (const f4*)(ev + base * 3);
        f4 e0 = __builtin_nontemporal_load(ep + 0);
        f4 e1 = __builtin_nontemporal_load(ep + 1);
        f4 e2 = __builtin_nontemporal_load(ep + 2);
        const float ev12[12] = {e0.x, e0.y, e0.z, e0.w,
                                e1.x, e1.y, e1.z, e1.w,
                                e2.x, e2.y, e2.z, e2.w};
        const float mm[4] = {m4.x, m4.y, m4.z, m4.w};
#pragma unroll
        for (int c = 0; c < 3; ++c) {
            Pack4 pk;
#pragma unroll
            for (int j = 0; j < 4; ++j) pk.sh[j] = f2bf(ev12[j * 3 + c] * mm[j]);
            *(uint64_t*)&A[c * 32 * APITCH + i * APITCH + jb] = pk.q;
        }
    };

    v8f acc[3][2] = {};
    stage(0, 0);

    for (int step = 0; step < N_ / 32; ++step) {
        const int jj  = step * 32;
        const int buf = step & 1;

        // ---- B fragments straight from global: 2 x 16B contiguous per frag
        BF16Frag bfr[2];
#pragma unroll
        for (int dn = 0; dn < 2; ++dn) {
            const int col = dgrp * 32 + dn * 16 + ln;
            const uint4* src = (const uint4*)(
                h_t + (((size_t)b * H_ + col) * N_ + jj + half * 16));
            uint4 q0 = src[0], q1 = src[1];
            bfr[dn].u[0] = q0.x; bfr[dn].u[1] = q0.y;
            bfr[dn].u[2] = q0.z; bfr[dn].u[3] = q0.w;
            bfr[dn].u[4] = q1.x; bfr[dn].u[5] = q1.y;
            bfr[dn].u[6] = q1.z; bfr[dn].u[7] = q1.w;
        }

        __syncthreads();   // current buf fully staged / prev reads done

        // ---- prefetch next A tile into the other buffer (overlaps WMMA)
        if (step + 1 < N_ / 32) stage(buf ^ 1, jj + 32);

        // ---- A fragments from LDS + 6 WMMAs (3 channels x 2 d-subtiles)
        const unsigned short* A = At + buf * (3 * 32 * APITCH);
        BF16Frag a[3];
        const int r = m_sub * 16 + ln;
#pragma unroll
        for (int c = 0; c < 3; ++c)
#pragma unroll
            for (int p = 0; p < 8; ++p) {
                int klocal = half * 16 + 2 * p;
                a[c].u[p] = *(const uint32_t*)&A[c * 32 * APITCH + r * APITCH + klocal];
            }
#pragma unroll
        for (int c = 0; c < 3; ++c)
#pragma unroll
            for (int dn = 0; dn < 2; ++dn)
                acc[c][dn] = __builtin_amdgcn_wmma_f32_16x16x32_bf16(
                    false, a[c].v, false, bfr[dn].v, (short)0, acc[c][dn], false, false);
    }

    // ---- epilogue: v[b,i,c,d], streaming stores (no reuse)
#pragma unroll
    for (int c = 0; c < 3; ++c) {
#pragma unroll
        for (int dn = 0; dn < 2; ++dn) {
            const int d = dgrp * 32 + dn * 16 + ln;
#pragma unroll
            for (int p = 0; p < 8; ++p) {
                int i = i0 + m_sub * 16 + p + half * 8;
                size_t idx = (((size_t)(b * N_ + i)) * 3 + c) * H_ + d;
                __builtin_nontemporal_store(acc[c][dn][p], out + idx);
            }
        }
    }
}

// ---------------------------------------------------------------------------
extern "C" void kernel_launch(void* const* d_in, const int* in_sizes, int n_in,
                              void* d_out, int out_size, void* d_ws, size_t ws_size,
                              hipStream_t stream) {
    const float* s    = (const float*)d_in[0];   // (B,N,H)
    const float* ev   = (const float*)d_in[1];   // (B,N,N,3)
    const float* mask = (const float*)d_in[2];   // (B,N,N,1)
    const float* W1   = (const float*)d_in[3];   // (H,H)
    const float* b1   = (const float*)d_in[4];   // (H,)
    unsigned short* h_t = (unsigned short*)d_ws; // h transposed [b][d][j], 4 MB

    lin1_ln_silu_kernel<<<dim3((B_ * N_) / 32), dim3(256), 0, stream>>>(s, W1, b1, h_t);
    cfconv_wmma_kernel<<<dim3(N_ / 32, B_), dim3(256), 0, stream>>>(
        ev, mask, h_t, (float*)d_out);
}
